// VLLMAttention_20151986553178
// MI455X (gfx1250) — compile-verified, matrix operand order checked by reading, not run on hardware
//
#include <hip/hip_runtime.h>
#include <hip/hip_bf16.h>
#include <math.h>

// Causal GQA flash-attention forward for MI455X (gfx1250, wave32).
// B=2, S=2048, H=32, HK=8, D=128. fp32 in/out, bf16 WMMA compute.
//
// Pipeline:
//   1. pre-pass A: K (fp32) -> Kb (bf16, same [b][s][hk][d] layout)   in d_ws
//   2. pre-pass B: V (fp32) -> Vt (bf16, transposed [b][hk][d][s])     in d_ws
//   3. main: per wave, 32 queries x full causal key range, flash softmax.
//      S^T = K·Q^T so softmax'd P^T (WMMA C/D layout) is directly the B
//      operand of O^T = V^T·P^T (C/D key striping == A/B k striping).
//      Longest query tiles are launched first (causal load balancing).
// Workspace requirement: 2 * (2*2048*8*128) * 2 bytes = 16.8 MB.

typedef __attribute__((ext_vector_type(16))) __bf16 v16bf;
typedef __attribute__((ext_vector_type(8)))  __bf16 v8bf;
typedef __attribute__((ext_vector_type(8)))  float  v8f;

#define WMMA_BF16(A, B, C) \
  __builtin_amdgcn_wmma_f32_16x16x32_bf16(false, (A), false, (B), (short)0, (C), false, false)

static __device__ __forceinline__ unsigned pack_bf16(float a, float b) {
  unsigned short x = __builtin_bit_cast(unsigned short, (__bf16)a);
  unsigned short y = __builtin_bit_cast(unsigned short, (__bf16)b);
  return (unsigned)x | ((unsigned)y << 16);
}

static __device__ __forceinline__ v16bf join16(v8bf lo, v8bf hi) {
  return __builtin_shufflevector(lo, hi, 0, 1, 2, 3, 4, 5, 6, 7,
                                         8, 9, 10, 11, 12, 13, 14, 15);
}

// fp32 row -> bf16 operand k-striping (elems 0..7 = base..base+7,
// 8..15 = base+16..base+23; base already includes +8 for upper lane half).
static __device__ __forceinline__ v16bf load_chunk_f32(const float* __restrict__ row,
                                                       int base, float scale) {
  const float4 f0 = *reinterpret_cast<const float4*>(row + base + 0);
  const float4 f1 = *reinterpret_cast<const float4*>(row + base + 4);
  const float4 f2 = *reinterpret_cast<const float4*>(row + base + 16);
  const float4 f3 = *reinterpret_cast<const float4*>(row + base + 20);
  v16bf m;
  m[0]  = (__bf16)(f0.x * scale); m[1]  = (__bf16)(f0.y * scale);
  m[2]  = (__bf16)(f0.z * scale); m[3]  = (__bf16)(f0.w * scale);
  m[4]  = (__bf16)(f1.x * scale); m[5]  = (__bf16)(f1.y * scale);
  m[6]  = (__bf16)(f1.z * scale); m[7]  = (__bf16)(f1.w * scale);
  m[8]  = (__bf16)(f2.x * scale); m[9]  = (__bf16)(f2.y * scale);
  m[10] = (__bf16)(f2.z * scale); m[11] = (__bf16)(f2.w * scale);
  m[12] = (__bf16)(f3.x * scale); m[13] = (__bf16)(f3.y * scale);
  m[14] = (__bf16)(f3.z * scale); m[15] = (__bf16)(f3.w * scale);
  return m;
}

// bf16 row -> operand: two contiguous 16-byte (8 x bf16) groups, no converts.
static __device__ __forceinline__ v16bf load_chunk_bf16(const __bf16* __restrict__ row,
                                                        int base) {
  v8bf lo = *reinterpret_cast<const v8bf*>(row + base);
  v8bf hi = *reinterpret_cast<const v8bf*>(row + base + 16);
  return join16(lo, hi);
}

// ---------------- pre-pass A: K fp32 -> bf16 (same layout) ----------------
__global__ __launch_bounds__(256)
void k_convert_bf16(const float* __restrict__ K, __bf16* __restrict__ Kb) {
  const size_t i = ((size_t)blockIdx.x * 256 + threadIdx.x) * 8;
  const float4 f0 = *reinterpret_cast<const float4*>(K + i);
  const float4 f1 = *reinterpret_cast<const float4*>(K + i + 4);
  uint4 u;
  u.x = pack_bf16(f0.x, f0.y);
  u.y = pack_bf16(f0.z, f0.w);
  u.z = pack_bf16(f1.x, f1.y);
  u.w = pack_bf16(f1.z, f1.w);
  *reinterpret_cast<uint4*>(Kb + i) = u;
}

// ------- pre-pass B: V fp32 [b][s][hk][d] -> Vt bf16 [b][hk][d][s] --------
__global__ __launch_bounds__(128)
void v_transpose_bf16(const float* __restrict__ V, __bf16* __restrict__ Vt) {
  constexpr int S = 2048, HK = 8, D = 128, ROWK = HK * D;
  const int d = threadIdx.x;                 // 0..127 (feature)
  int bid = blockIdx.x;
  const int sb = bid & 31; bid >>= 5;        // 32 key-blocks of 64
  const int hk = bid & 7;  bid >>= 3;
  const int b  = bid;
  const float* src = V + ((size_t)b * S + (size_t)sb * 64) * ROWK + (size_t)hk * D + d;
  __bf16* dst = Vt + (((size_t)b * HK + hk) * D + d) * S + (size_t)sb * 64;
#pragma unroll
  for (int s0 = 0; s0 < 64; s0 += 8) {
    uint4 u;
    u.x = pack_bf16(src[(size_t)(s0 + 0) * ROWK], src[(size_t)(s0 + 1) * ROWK]);
    u.y = pack_bf16(src[(size_t)(s0 + 2) * ROWK], src[(size_t)(s0 + 3) * ROWK]);
    u.z = pack_bf16(src[(size_t)(s0 + 4) * ROWK], src[(size_t)(s0 + 5) * ROWK]);
    u.w = pack_bf16(src[(size_t)(s0 + 6) * ROWK], src[(size_t)(s0 + 7) * ROWK]);
    *reinterpret_cast<uint4*>(dst + s0) = u;
  }
}

// ----------------------------- main kernel --------------------------------
__global__ __launch_bounds__(32)
void gqa_flash_fwd(const float* __restrict__ Q, const __bf16* __restrict__ Kb,
                   const __bf16* __restrict__ Vt, float* __restrict__ out) {
  constexpr int S = 2048, H = 32, HK = 8, D = 128, ROWK = HK * D;
  constexpr float NEG = -3.0e38f;

  const int lane = threadIdx.x;
  const int col  = lane & 15;                // N index (query) / feature-in-chunk
  const int hi   = lane >> 4;                // lane half -> +8 in k-striping

  // longest causal wedges first: qt descending in the high bits of blockIdx
  const int bid = blockIdx.x;
  const int qt  = 63 - (bid >> 6);           // 64 query tiles of 32
  const int bh  = bid & 63;
  const int h   = bh & 31;                   // 32 query heads
  const int b   = bh >> 5;                   // 2 batches
  const int hk  = h >> 2;                    // G = 4
  const int q_base = qt * 32;

  const float qscale = 1.44269504088896f * 0.088388347648318447f; // log2e/sqrt(128)

  // ---- preload two 16-query Q blocks as B operands
  const float* qrow0 = Q + (((size_t)b * S + q_base + col) * H + h) * D;
  const float* qrow1 = qrow0 + (size_t)16 * H * D;
  v16bf qb0[4], qb1[4];
#pragma unroll
  for (int c = 0; c < 4; ++c) {
    qb0[c] = load_chunk_f32(qrow0, c * 32 + hi * 8, qscale);
    qb1[c] = load_chunk_f32(qrow1, c * 32 + hi * 8, qscale);
  }

  const v8f vzero = (v8f){0.f, 0.f, 0.f, 0.f, 0.f, 0.f, 0.f, 0.f};
  v8f o0[8], o1[8];
#pragma unroll
  for (int c = 0; c < 8; ++c) { o0[c] = vzero; o1[c] = vzero; }
  float m0 = NEG, l0 = 0.0f, m1 = NEG, l1 = 0.0f;

  const __bf16* kbat = Kb + (size_t)b * S * ROWK + (size_t)hk * D;
  const __bf16* vbat = Vt + (((size_t)b * HK + hk) * (size_t)D) * S;

  const int ntiles = qt + 1;                 // keys 0 .. q_base+31
  for (int t = 0; t < ntiles; ++t) {
    const int kbase = t * 32;

    if (t + 1 < ntiles) {                    // global_prefetch_b8 next tile
      __builtin_prefetch(kbat + (size_t)(kbase + 32 + col) * ROWK, 0, 3);
      __builtin_prefetch(vbat + (size_t)(lane * 4) * S + kbase + 32, 0, 3);
    }

    // ---- S^T = K · Q^T : keys 0-15 / 16-31 x query blocks 0 / 1
    const __bf16* krow0 = kbat + (size_t)(kbase + col) * ROWK;   // M = key = col
    const __bf16* krow1 = krow0 + (size_t)16 * ROWK;
    v8f s00 = vzero, s01 = vzero, s10 = vzero, s11 = vzero;
#pragma unroll
    for (int c = 0; c < 4; ++c) {
      const int base = c * 32 + hi * 8;
      v16bf ka0 = load_chunk_bf16(krow0, base);
      v16bf ka1 = load_chunk_bf16(krow1, base);
      s00 = WMMA_BF16(ka0, qb0[c], s00);
      s10 = WMMA_BF16(ka1, qb0[c], s10);
      s01 = WMMA_BF16(ka0, qb1[c], s01);
      s11 = WMMA_BF16(ka1, qb1[c], s11);
    }

    // ---- causal mask (only the diagonal tile per block needs it)
    const int qg0 = q_base + col, qg1 = qg0 + 16;
    if (kbase + 31 > q_base) {
#pragma unroll
      for (int r = 0; r < 8; ++r) {
        const int key0 = kbase + r + 8 * hi;
        if (key0 > qg0)      s00[r] = NEG;
        if (key0 + 16 > qg0) s10[r] = NEG;
      }
    }
    if (kbase + 31 > q_base + 16) {
#pragma unroll
      for (int r = 0; r < 8; ++r) {
        const int key0 = kbase + r + 8 * hi;
        if (key0 > qg1)      s01[r] = NEG;
        if (key0 + 16 > qg1) s11[r] = NEG;
      }
    }

    // ---- online softmax, block 0
    float tmax = NEG;
#pragma unroll
    for (int r = 0; r < 8; ++r) tmax = fmaxf(tmax, fmaxf(s00[r], s10[r]));
    tmax = fmaxf(tmax, __shfl_xor(tmax, 16, 32));
    const float mn0 = fmaxf(m0, tmax);
    const float alpha0 = __builtin_exp2f(m0 - mn0);
    float ps0 = 0.0f;
    v16bf pb0;
#pragma unroll
    for (int r = 0; r < 8; ++r) {
      const float p0 = __builtin_exp2f(s00[r] - mn0);
      const float p1 = __builtin_exp2f(s10[r] - mn0);
      ps0 += p0 + p1;
      pb0[r] = (__bf16)p0;  pb0[r + 8] = (__bf16)p1;
    }
    ps0 += __shfl_xor(ps0, 16, 32);
    l0 = l0 * alpha0 + ps0;  m0 = mn0;

    // ---- online softmax, block 1
    tmax = NEG;
#pragma unroll
    for (int r = 0; r < 8; ++r) tmax = fmaxf(tmax, fmaxf(s01[r], s11[r]));
    tmax = fmaxf(tmax, __shfl_xor(tmax, 16, 32));
    const float mn1 = fmaxf(m1, tmax);
    const float alpha1 = __builtin_exp2f(m1 - mn1);
    float ps1 = 0.0f;
    v16bf pb1;
#pragma unroll
    for (int r = 0; r < 8; ++r) {
      const float p0 = __builtin_exp2f(s01[r] - mn1);
      const float p1 = __builtin_exp2f(s11[r] - mn1);
      ps1 += p0 + p1;
      pb1[r] = (__bf16)p0;  pb1[r + 8] = (__bf16)p1;
    }
    ps1 += __shfl_xor(ps1, 16, 32);
    l1 = l1 * alpha1 + ps1;  m1 = mn1;

    // ---- O^T += V^T · P^T : Vt rows are key-contiguous -> 2 b128 loads/chunk
#pragma unroll
    for (int c = 0; c < 8; ++c) {
      const __bf16* vrow = vbat + (size_t)(c * 16 + col) * S + kbase + 8 * hi;
      v16bf va = join16(*reinterpret_cast<const v8bf*>(vrow),
                        *reinterpret_cast<const v8bf*>(vrow + 16));
#pragma unroll
      for (int r = 0; r < 8; ++r) { o0[c][r] *= alpha0; o1[c][r] *= alpha1; }
      o0[c] = WMMA_BF16(va, pb0, o0[c]);
      o1[c] = WMMA_BF16(va, pb1, o1[c]);
    }
  }

  // ---- epilogue: O = O^T / l
  const float il0 = 1.0f / l0, il1 = 1.0f / l1;
  float* orow0 = out + ((size_t)b * S + q_base + col) * (H * D) + (size_t)h * D;
  float* orow1 = orow0 + (size_t)16 * H * D;
#pragma unroll
  for (int c = 0; c < 8; ++c) {
#pragma unroll
    for (int r = 0; r < 8; ++r) {
      orow0[c * 16 + r + 8 * hi] = o0[c][r] * il0;
      orow1[c * 16 + r + 8 * hi] = o1[c][r] * il1;
    }
  }
}

extern "C" void kernel_launch(void* const* d_in, const int* in_sizes, int n_in,
                              void* d_out, int out_size, void* d_ws, size_t ws_size,
                              hipStream_t stream) {
  (void)in_sizes; (void)n_in; (void)out_size; (void)ws_size;
  const float* q = (const float*)d_in[0];
  const float* k = (const float*)d_in[1];
  const float* v = (const float*)d_in[2];
  float* out = (float*)d_out;

  constexpr size_t NKV = (size_t)2 * 2048 * 8 * 128;   // 4,194,304 elements
  __bf16* kbf = (__bf16*)d_ws;                          // 8.4 MB
  __bf16* vt  = kbf + NKV;                              // 8.4 MB (total 16.8 MB)

  // pre-pass: bf16 convert K, transpose+convert V (same stream => ordered)
  k_convert_bf16<<<dim3((unsigned)(NKV / (256 * 8))), dim3(256), 0, stream>>>(k, kbf);
  v_transpose_bf16<<<dim3(512), dim3(128), 0, stream>>>(v, vt);

  // main: B * H * (S/32) = 2 * 32 * 64 = 4096 waves, one wave32 per block
  gqa_flash_fwd<<<dim3(4096), dim3(32), 0, stream>>>(q, kbf, vt, out);
}
